// MultiHeadAttention_91061896610551
// MI455X (gfx1250) — matrix operand order repaired and checked by reading, not compiled, not run on hardware
//
#include <hip/hip_runtime.h>
#include <hip/hip_bf16.h>
#include <cstdint>

// Problem constants (match reference)
#define E_DIM 1024
#define NHEAD 16
#define HDIM 64
#define BATCH 2
#define SEQQ 2048
#define SEQK 2048
#define MROWS (BATCH * SEQQ)   // 4096

typedef __attribute__((ext_vector_type(16))) __bf16        v16bf;
typedef __attribute__((ext_vector_type(8)))  float         v8f;
typedef __attribute__((ext_vector_type(4)))  unsigned int  u32x4;

// GCC-style vector matching the async-LDS builtin's parameter type exactly:
// 'int __attribute__((vector_size(16)))' in AS1 (global) / AS3 (LDS).
typedef int v4i __attribute__((vector_size(16)));
typedef __attribute__((address_space(1))) v4i global_v4i;
typedef __attribute__((address_space(3))) v4i local_v4i;

// POD union for building WMMA bf16 fragments out of two 16-byte LDS loads.
union Frag16 { u32x4 u[2]; v16bf v; };

static __device__ __forceinline__ v8f wmma_bf16(v16bf a, v16bf b, v8f c) {
  return __builtin_amdgcn_wmma_f32_16x16x32_bf16(false, a, false, b, (short)0, c,
                                                 false, false);
}

// ---------------------------------------------------------------------------
// Async global->LDS 16B copy (CDNA5 GLOBAL_LOAD_ASYNC_TO_LDS_B128, ASYNCcnt).
// Generic->AS conversion done numerically via inttoptr: global keeps the full
// 64-bit flat address; for the LDS aperture the low 32 bits of a flat address
// are the LDS byte address (ISA 10.2), so u32-truncate for the AS3 pointer.
// Falls back to a synchronous copy if the builtin is not declared.
// ---------------------------------------------------------------------------
static __device__ __forceinline__ void async_ld_b128(const void* g, void* l) {
#if __has_builtin(__builtin_amdgcn_global_load_async_to_lds_b128)
  __builtin_amdgcn_global_load_async_to_lds_b128(
      (global_v4i*)(uintptr_t)g,
      (local_v4i*)(uintptr_t)(unsigned)(uintptr_t)l,
      0, 0);
#else
  *(u32x4*)l = *(const u32x4*)g;
#endif
}

static __device__ __forceinline__ void wait_async0() {
#if __has_builtin(__builtin_amdgcn_s_wait_asynccnt)
  __builtin_amdgcn_s_wait_asynccnt(0);
#elif __has_builtin(__builtin_amdgcn_global_load_async_to_lds_b128)
  asm volatile("s_wait_asynccnt 0x0" ::: "memory");
#endif
}

// ---------------------------------------------------------------------------
// fp32 -> bf16 elementwise conversion
// ---------------------------------------------------------------------------
__global__ __launch_bounds__(256) void cvt_f32_bf16(const float* __restrict__ in,
                                                    __bf16* __restrict__ out,
                                                    int n) {
  int i = blockIdx.x * blockDim.x + threadIdx.x;
  int stride = gridDim.x * blockDim.x;
  for (; i < n; i += stride) out[i] = (__bf16)in[i];
}

// ---------------------------------------------------------------------------
// C[M,N] = A[M,K] @ W[K,N] + bias   (A,W bf16 row-major; out bf16 or f32)
// Block = 256 threads = 8 waves; 128x128 output tile per workgroup.
// K staged 64 at a time via async-LDS copies; wave (wm,wn) in a 2x4 grid
// computes a 64x32 sub-tile = 4x2 WMMA 16x16 accumulators (16 WMMAs / stage).
// ---------------------------------------------------------------------------
template <bool OUT_BF16>
__global__ __launch_bounds__(256) void gemm_bias_bf16(
    const __bf16* __restrict__ A, const __bf16* __restrict__ W,
    const float* __restrict__ bias, void* __restrict__ Cout,
    int Mtot, int N, int K) {
  __shared__ __bf16 lA[128][64];   // A tile  (m x k)   16 KB
  __shared__ __bf16 lB[64][128];   // W tile  (k x n)   16 KB

  const int t    = threadIdx.x;
  const int lane = t & 31;
  const int wave = t >> 5;
  const int wm   = wave >> 2;      // 0..1
  const int wn   = wave & 3;       // 0..3
  const int g    = lane >> 4;      // lane half
  const int lr   = lane & 15;
  const int m0   = blockIdx.x * 128;
  const int n0   = blockIdx.y * 128;

  v8f acc[4][2] = {};

  for (int kt = 0; kt < K; kt += 64) {
    __syncthreads();  // previous stage's LDS reads are consumed before rewrite
    {  // stage A tile 128x64: thread -> (row, 32-col half), 4 async b128
      const int r = t >> 1, c = (t & 1) * 32;
      const __bf16* src = A + (size_t)(m0 + r) * K + kt + c;
#pragma unroll
      for (int u = 0; u < 4; ++u) async_ld_b128(src + u * 8, &lA[r][c + u * 8]);
    }
    {  // stage W tile 64x128 row-major: thread -> (k row, 32-col chunk)
      const int k2 = t >> 2, c = (t & 3) * 32;
      const __bf16* src = W + (size_t)(kt + k2) * N + n0 + c;
#pragma unroll
      for (int u = 0; u < 4; ++u) async_ld_b128(src + u * 8, &lB[k2][c + u * 8]);
    }
    wait_async0();
    __syncthreads();

#pragma unroll
    for (int ks = 0; ks < 2; ++ks) {  // two K=32 WMMA steps per stage
      v16bf af[4], bfr[2];
#pragma unroll
      for (int i = 0; i < 4; ++i) {
        // A fragment: lane = M row, elements K = ks*32 + {8g..8g+7, 8g+16..+23}
        Frag16 f;
        const __bf16* p = &lA[wm * 64 + i * 16 + lr][ks * 32 + g * 8];
        f.u[0] = *(const u32x4*)p;
        f.u[1] = *(const u32x4*)(p + 16);
        af[i] = f.v;
      }
#pragma unroll
      for (int j = 0; j < 2; ++j) {
        // B fragment: lane = N column, elements K = ks*32 + {8g..8g+7, 8g+16..+23}
        v16bf f;
#pragma unroll
        for (int s = 0; s < 16; ++s)
          f[s] = lB[ks * 32 + g * 8 + (s & 7) + (s >> 3) * 16][wn * 32 + j * 16 + lr];
        bfr[j] = f;
      }
#pragma unroll
      for (int i = 0; i < 4; ++i)
#pragma unroll
        for (int j = 0; j < 2; ++j)
          acc[i][j] = wmma_bf16(af[i], bfr[j], acc[i][j]);
    }
  }

  // Epilogue: C layout -> VGPR j holds row (j + 8g), lane lr holds column.
#pragma unroll
  for (int j = 0; j < 2; ++j) {
    const int col = n0 + wn * 32 + j * 16 + lr;
    const float bv = bias[col];
#pragma unroll
    for (int i = 0; i < 4; ++i) {
#pragma unroll
      for (int e = 0; e < 8; ++e) {
        const int row = m0 + wm * 64 + i * 16 + e + g * 8;
        const float o = acc[i][j][e] + bv;
        if (OUT_BF16)
          ((__bf16*)Cout)[(size_t)row * N + col] = (__bf16)o;
        else
          ((float*)Cout)[(size_t)row * N + col] = o;
      }
    }
  }
}

// ---------------------------------------------------------------------------
// Flash attention over projected Q/K/V (bf16, layout [b*seq][h*64+d]).
// grid = (SEQQ/128, BATCH*NHEAD); block = 256 (8 waves).
// Each wave owns 16 query rows; all waves share LDS-staged 32-key K/V blocks
// (V via async-LDS copy, K transposed manually). Online softmax in fp32;
// P bounced through a wave-private LDS tile to convert C->A layout.
// ---------------------------------------------------------------------------
__global__ __launch_bounds__(256) void flash_attn_bf16(
    const __bf16* __restrict__ Qp, const __bf16* __restrict__ Kp,
    const __bf16* __restrict__ Vp, __bf16* __restrict__ O) {
  __shared__ __bf16 lKt[64][32];       // K block transposed: [d][ks]
  __shared__ __bf16 lV[32][64];        // V block row-major:  [ks][d]
  __shared__ __bf16 lP[8][16][32];     // per-wave P bounce tile

  const int t    = threadIdx.x;
  const int lane = t & 31;
  const int wave = t >> 5;
  const int g    = lane >> 4;
  const int lr   = lane & 15;
  const int qb   = blockIdx.x;
  const int b    = blockIdx.y >> 4;
  const int h    = blockIdx.y & 15;
  const size_t base = ((size_t)b * SEQQ) * E_DIM + (size_t)h * HDIM;
  const int qrow = qb * 128 + wave * 16 + lr;   // this lane's A-fragment M row

  // Q fragments (fixed across key loop): d = s*32 + {8g..8g+7, 8g+16..8g+23}
  v16bf qf[2];
#pragma unroll
  for (int s = 0; s < 2; ++s) {
    Frag16 f;
    const __bf16* p = Qp + base + (size_t)qrow * E_DIM + s * 32 + g * 8;
    f.u[0] = *(const u32x4*)p;
    f.u[1] = *(const u32x4*)(p + 16);
    qf[s] = f.v;
  }

  v8f acc[4] = {};
  float mrow[8], lsum[8];
#pragma unroll
  for (int e = 0; e < 8; ++e) { mrow[e] = -1e30f; lsum[e] = 0.f; }
  const float scale = 0.125f;  // 1/sqrt(64)

  for (int kb = 0; kb < SEQK / 32; ++kb) {
    __syncthreads();
    {  // cooperative staging: K transposed (manual), V row-major (async)
      const int ks = t >> 3, d0 = (t & 7) * 8;
      const size_t roff = base + (size_t)(kb * 32 + ks) * E_DIM + d0;
      u32x4 kv = *(const u32x4*)(Kp + roff);
      const __bf16* ke = (const __bf16*)&kv;
#pragma unroll
      for (int i = 0; i < 8; ++i) lKt[d0 + i][ks] = ke[i];
      async_ld_b128(Vp + roff, &lV[ks][d0]);
      if (kb + 1 < SEQK / 32) {  // prefetch next block -> global_prefetch_b8
        __builtin_prefetch(Kp + roff + 32 * E_DIM, 0, 1);
        __builtin_prefetch(Vp + roff + 32 * E_DIM, 0, 1);
      }
    }
    wait_async0();
    __syncthreads();

    // S = Q @ K^T : two 16x16 score tiles (key cols 0..15, 16..31)
    v8f sv[2];
#pragma unroll
    for (int tt = 0; tt < 2; ++tt) {
      v8f z = {};
#pragma unroll
      for (int ds = 0; ds < 2; ++ds) {
        v16bf f;  // B fragment: lane = key column, elems d = ds*32 + {8g..8g+7, 8g+16..+23}
#pragma unroll
        for (int s = 0; s < 16; ++s)
          f[s] = lKt[ds * 32 + g * 8 + (s & 7) + (s >> 3) * 16][tt * 16 + lr];
        z = wmma_bf16(qf[ds], f, z);
      }
      sv[tt] = z;
    }

    // Online softmax (row stats live per accum-element, reduced over 16 lanes)
    float sc0[8], sc1[8], pm[8];
#pragma unroll
    for (int e = 0; e < 8; ++e) {
      sc0[e] = sv[0][e] * scale;
      sc1[e] = sv[1][e] * scale;
      pm[e]  = fmaxf(sc0[e], sc1[e]);
    }
#pragma unroll
    for (int off = 1; off <= 8; off <<= 1) {
#pragma unroll
      for (int e = 0; e < 8; ++e)
        pm[e] = fmaxf(pm[e], __shfl_xor(pm[e], off, 32));
    }
    float alpha[8], rs[8];
#pragma unroll
    for (int e = 0; e < 8; ++e) {
      const float mn = fmaxf(mrow[e], pm[e]);
      alpha[e] = __expf(mrow[e] - mn);
      mrow[e]  = mn;
      sc0[e]   = __expf(sc0[e] - mn);
      sc1[e]   = __expf(sc1[e] - mn);
      rs[e]    = sc0[e] + sc1[e];
    }
#pragma unroll
    for (int off = 1; off <= 8; off <<= 1) {
#pragma unroll
      for (int e = 0; e < 8; ++e) rs[e] += __shfl_xor(rs[e], off, 32);
    }
#pragma unroll
    for (int e = 0; e < 8; ++e) lsum[e] = lsum[e] * alpha[e] + rs[e];
#pragma unroll
    for (int t4 = 0; t4 < 4; ++t4)
#pragma unroll
      for (int e = 0; e < 8; ++e) acc[t4][e] *= alpha[e];

    // P: C-layout -> LDS -> A-layout fragment (wave-private; DS ops in-order)
#pragma unroll
    for (int e = 0; e < 8; ++e) {
      lP[wave][e + g * 8][lr]      = (__bf16)sc0[e];
      lP[wave][e + g * 8][16 + lr] = (__bf16)sc1[e];
    }
    Frag16 pf;
    {
      const __bf16* p = &lP[wave][lr][g * 8];
      pf.u[0] = *(const u32x4*)p;
      pf.u[1] = *(const u32x4*)(p + 16);
    }
    // acc += P @ V : 4 output d-tiles; V B-fragment lane = d col, elems = ks (table order)
#pragma unroll
    for (int t4 = 0; t4 < 4; ++t4) {
      v16bf vf;
#pragma unroll
      for (int s = 0; s < 16; ++s)
        vf[s] = lV[g * 8 + (s & 7) + (s >> 3) * 16][t4 * 16 + lr];
      acc[t4] = wmma_bf16(pf.v, vf, acc[t4]);
    }
  }

  // Normalize and store attention output (bf16) in [b*seq][h*64+d] layout
#pragma unroll
  for (int t4 = 0; t4 < 4; ++t4)
#pragma unroll
    for (int e = 0; e < 8; ++e) {
      const int row = qb * 128 + wave * 16 + e + g * 8;
      const float o = acc[t4][e] / lsum[e];
      O[((size_t)b * SEQQ + row) * E_DIM + (size_t)h * HDIM + t4 * 16 + lr] =
          (__bf16)o;
    }
}

// ---------------------------------------------------------------------------
extern "C" void kernel_launch(void* const* d_in, const int* in_sizes, int n_in,
                              void* d_out, int out_size, void* d_ws,
                              size_t ws_size, hipStream_t stream) {
  const float* q  = (const float*)d_in[0];
  const float* k  = (const float*)d_in[1];
  const float* v  = (const float*)d_in[2];
  const float* Wq = (const float*)d_in[3];
  const float* bq = (const float*)d_in[4];
  const float* Wk = (const float*)d_in[5];
  const float* bk = (const float*)d_in[6];
  const float* Wv = (const float*)d_in[7];
  const float* bv = (const float*)d_in[8];
  const float* Wo = (const float*)d_in[9];
  const float* bo = (const float*)d_in[10];
  float* out = (float*)d_out;

  // Workspace layout (all bf16): 3 inputs + 4 weights + Q/K/V proj + attn out
  // = 3*8 + 4*2 + 3*8 + 8 = 64 MiB.
  char* ws = (char*)d_ws;
  size_t off = 0;
  auto carve = [&](size_t elems) {
    __bf16* p = (__bf16*)(ws + off);
    off += elems * sizeof(__bf16);
    return p;
  };
  const size_t ME = (size_t)MROWS * E_DIM;
  const size_t EE = (size_t)E_DIM * E_DIM;
  __bf16* xq  = carve(ME);
  __bf16* xk  = carve(ME);
  __bf16* xv  = carve(ME);
  __bf16* wqb = carve(EE);
  __bf16* wkb = carve(EE);
  __bf16* wvb = carve(EE);
  __bf16* wob = carve(EE);
  __bf16* Qp  = carve(ME);
  __bf16* Kp  = carve(ME);
  __bf16* Vp  = carve(ME);
  __bf16* Ob  = carve(ME);
  (void)ws_size; (void)in_sizes; (void)n_in; (void)out_size;

  auto cvt = [&](const float* src, __bf16* dst, size_t n) {
    const int blocks = (int)((n + 1023) / 1024);
    cvt_f32_bf16<<<blocks, 256, 0, stream>>>(src, dst, (int)n);
  };
  cvt(q, xq, ME);
  cvt(k, xk, ME);
  cvt(v, xv, ME);
  cvt(Wq, wqb, EE);
  cvt(Wk, wkb, EE);
  cvt(Wv, wvb, EE);
  cvt(Wo, wob, EE);

  dim3 gg(MROWS / 128, E_DIM / 128);
  gemm_bias_bf16<true><<<gg, 256, 0, stream>>>(xq, wqb, bq, Qp, MROWS, E_DIM, E_DIM);
  gemm_bias_bf16<true><<<gg, 256, 0, stream>>>(xk, wkb, bk, Kp, MROWS, E_DIM, E_DIM);
  gemm_bias_bf16<true><<<gg, 256, 0, stream>>>(xv, wvb, bv, Vp, MROWS, E_DIM, E_DIM);

  dim3 ga(SEQQ / 128, BATCH * NHEAD);
  flash_attn_bf16<<<ga, 256, 0, stream>>>(Qp, Kp, Vp, Ob);

  gemm_bias_bf16<false><<<gg, 256, 0, stream>>>(Ob, wob, bo, out, MROWS, E_DIM, E_DIM);
}
